// AtomicMultiHeadAttention_24927990186257
// MI455X (gfx1250) — matrix operand order repaired and checked by reading, not compiled, not run on hardware
//
#include <hip/hip_runtime.h>
#include <hip/hip_bf16.h>

// ---------------------------------------------------------------------------
// AtomicMultiHeadAttention for MI455X (gfx1250, wave32)
//   - WMMA bf16 16x16x32 for all matrix math (fp32 accum)
//   - GLOBAL_LOAD_ASYNC_TO_LDS_B128 (ASYNCcnt) for per-wave B-tile staging
//   - Tensor Data Mover (TENSORcnt) for block-shared W1/W2/K-row staging
// B=2, S=512, D=256, H=8, DP=32
// ---------------------------------------------------------------------------

typedef __attribute__((ext_vector_type(16))) __bf16 v16bf;
typedef __attribute__((ext_vector_type(8)))  __bf16 v8bf;
typedef __attribute__((ext_vector_type(8)))  float  v8f;
typedef __attribute__((ext_vector_type(4)))  unsigned int u32x4;
typedef __attribute__((ext_vector_type(8)))  int i32x8;
typedef __attribute__((ext_vector_type(4)))  int i32x4;

#define BB  2
#define SS  512
#define DDm 256
#define HHn 8
#define DPP 32

union V8F   { v8f  v; float  f[8]; };
union V8BF  { v8bf v; __bf16 e[8]; };
union V16BF { v16bf v; v8bf h[2]; };

// ---- low-level CDNA5 helpers ----------------------------------------------

__device__ __forceinline__ unsigned int lds_off(const void* p) {
    // flat LDS address = {SHARED_BASE aperture, 32-bit LDS byte offset}
    return (unsigned int)(unsigned long long)p;
}

__device__ __forceinline__ void async_copy_b128(unsigned int lds_dst, const void* gsrc) {
    unsigned long long ga = (unsigned long long)gsrc;
    asm volatile("global_load_async_to_lds_b128 %0, %1, off"
                 :: "v"(lds_dst), "v"(ga) : "memory");
}

__device__ __forceinline__ void wait_async0() {
    asm volatile("s_wait_asynccnt 0x0" ::: "memory");
}

__device__ __forceinline__ void wait_ds0() {
    asm volatile("s_wait_dscnt 0x0" ::: "memory");
}

// TDM: 1-D linear copy of nelem bf16 elements global -> LDS (one per wave).
// D# per CDNA5 ISA 8.3/8.4: count=1, type=2 (image), data_size=1 (2 bytes),
// tile_dim0 = nelem, tensor_dim0 = tensor_dim0_stride = nelem, 1-D.
__device__ __forceinline__ void tdm_copy_1d(unsigned int lds_dst, const void* gsrc,
                                            unsigned int nelem) {
    unsigned long long ga = (unsigned long long)gsrc;
    u32x4 g0;
    g0[0] = 1u;                                            // count=1, user desc
    g0[1] = lds_dst;                                       // lds_addr (bytes)
    g0[2] = (unsigned int)(ga & 0xffffffffu);              // global_addr[31:0]
    g0[3] = (unsigned int)((ga >> 32) & 0x01ffffffu) | (2u << 30); // [56:32] | type=2
    i32x8 g1;
    g1[0] = (int)(1u << 16);                               // data_size=1 (2B)
    g1[1] = (int)((nelem & 0xffffu) << 16);                // tensor_dim0[15:0]
    g1[2] = (int)(nelem >> 16);                            // tensor_dim0[31:16]
    g1[3] = (int)((nelem & 0xffffu) << 16);                // tile_dim0
    g1[4] = 0;
    g1[5] = (int)nelem;                                    // tensor_dim0_stride[31:0]
    g1[6] = 0;
    g1[7] = 0;
    i32x4 z4; z4[0] = 0; z4[1] = 0; z4[2] = 0; z4[3] = 0;
    i32x8 z8; z8[0] = 0; z8[1] = 0; z8[2] = 0; z8[3] = 0;
    z8[4] = 0; z8[5] = 0; z8[6] = 0; z8[7] = 0;
    // 6-arg form (this toolchain): (g0, g1, g2, g3, g4, cpol)
    __builtin_amdgcn_tensor_load_to_lds(g0, g1, z4, z4, z8, 0);
}

__device__ __forceinline__ void wait_tensor0() {
    __builtin_amdgcn_s_wait_tensorcnt(0);
}

// ---- WMMA fragment helpers (layouts per CDNA5 ISA 7.12.2) -----------------

// A fragment: 16x32 bf16, row-major LDS tile As[16][32].
__device__ __forceinline__ v16bf load_a_frag(const __bf16* As, int lane) {
    int r    = lane & 15;
    int base = (lane >> 4) * 8;
    V16BF u;
    u.h[0] = *(const v8bf*)(As + r * 32 + base);
    u.h[1] = *(const v8bf*)(As + r * 32 + base + 16);
    return u.v;
}

// B fragment: 32x16 bf16 stored transposed in LDS as Bs[n][k].
__device__ __forceinline__ v16bf load_b_frag(const __bf16* Bs, int lane) {
    int c  = lane & 15;
    int kb = (lane >> 4) * 8;
    V16BF u;
    u.h[0] = *(const v8bf*)(Bs + c * 32 + kb);
    u.h[1] = *(const v8bf*)(Bs + c * 32 + kb + 16);
    return u.v;
}

__device__ __forceinline__ v8f wmma_bf16(v16bf a, v16bf b, v8f c) {
    return __builtin_amdgcn_wmma_f32_16x16x32_bf16(false, a, false, b,
                                                   (short)0, c, false, false);
}

// ---- A-tile staging: 16x32 fp32 chunk -> bf16 LDS (per wave) --------------
__device__ __forceinline__ void stage_a(const float* A, int lda, int m0, int k0,
                                        __bf16* As, int lane) {
    int r  = lane >> 1;
    int ch = (lane & 1) * 16;
    const float* p = A + (m0 + r) * lda + k0 + ch;
    float4 f0 = *(const float4*)(p);
    float4 f1 = *(const float4*)(p + 4);
    float4 f2 = *(const float4*)(p + 8);
    float4 f3 = *(const float4*)(p + 12);
    V8BF x0, x1;
    x0.e[0] = (__bf16)f0.x; x0.e[1] = (__bf16)f0.y; x0.e[2] = (__bf16)f0.z; x0.e[3] = (__bf16)f0.w;
    x0.e[4] = (__bf16)f1.x; x0.e[5] = (__bf16)f1.y; x0.e[6] = (__bf16)f1.z; x0.e[7] = (__bf16)f1.w;
    x1.e[0] = (__bf16)f2.x; x1.e[1] = (__bf16)f2.y; x1.e[2] = (__bf16)f2.z; x1.e[3] = (__bf16)f2.w;
    x1.e[4] = (__bf16)f3.x; x1.e[5] = (__bf16)f3.y; x1.e[6] = (__bf16)f3.z; x1.e[7] = (__bf16)f3.w;
    __bf16* d = As + r * 32 + ch;
    *(v8bf*)(d)     = x0.v;
    *(v8bf*)(d + 8) = x1.v;
}

// ---- wave-level 16x16 GEMM: A fp32 (cvt), B pre-transposed bf16 (async) ---
// BT layout: BT[n_global * ldk + k_global]  (column-major original B)
__device__ __forceinline__ void wave_gemm_bfB(const float* A, int lda, int m0,
                                              const __bf16* BT, int ldk, int n0, int K,
                                              __bf16* As, __bf16* Bs, int lane,
                                              float out[8]) {
    V8F acc;
#pragma unroll
    for (int i = 0; i < 8; ++i) acc.f[i] = 0.0f;
    unsigned int bsOff = lds_off(Bs);
    int n  = lane >> 1;
    int sg = lane & 1;
    for (int k0 = 0; k0 < K; k0 += 32) {
        wait_ds0();   // prior fragment loads complete before LDS overwrite
        // B-tile: per-lane async copies (2 x 16B), covers k = sg*8..+7, sg*8+16..+7
        const __bf16* src = BT + (size_t)(n0 + n) * ldk + (k0 + sg * 8);
        unsigned int dst = bsOff + (unsigned int)(n * 64 + sg * 16);
        async_copy_b128(dst,        src);
        async_copy_b128(dst + 32u,  src + 16);
        // A-tile: fp32 -> bf16 through VGPRs
        stage_a(A, lda, m0, k0, As, lane);
        wait_async0();
        acc.v = wmma_bf16(load_a_frag(As, lane), load_b_frag(Bs, lane), acc.v);
    }
#pragma unroll
    for (int i = 0; i < 8; ++i) out[i] = acc.f[i];
}

// ---------------------------------------------------------------------------
// Kernel 0: weight prep — bf16 + transpose so all B-tiles are linear copies.
//   WxT[c*256+k] = Wx[k*256+c]; W1T[c*32+k] = W1[k*32+c]; W2T[c*32+k] = W2[k*256+c]
// ---------------------------------------------------------------------------
__global__ void __launch_bounds__(256)
k_prep(const float* Wq, const float* Wk, const float* Wv, const float* Wo,
       const float* W1, const float* W2,
       __bf16* WqT, __bf16* WkT, __bf16* WvT, __bf16* WoT,
       __bf16* W1T, __bf16* W2T) {
    int tid = threadIdx.x;
    int y = blockIdx.y;
    if (y < 4) {
        const float* W = (y == 0) ? Wq : (y == 1) ? Wk : (y == 2) ? Wv : Wo;
        __bf16* WT     = (y == 0) ? WqT : (y == 1) ? WkT : (y == 2) ? WvT : WoT;
        int base = blockIdx.x * 1024;
#pragma unroll
        for (int u = 0; u < 4; ++u) {
            int g = base + tid * 4 + u;          // 0..65535
            int c = g >> 8, k = g & 255;
            WT[g] = (__bf16)W[k * 256 + c];
        }
    } else if (y == 4) {
        if (blockIdx.x == 0) {
#pragma unroll
            for (int u = 0; u < 4; ++u) {
                int g = tid * 4 + u;             // 0..1023
                int c = g >> 5, k = g & 31;
                W1T[g] = (__bf16)W1[k * 32 + c];
            }
        }
    } else {
        if (blockIdx.x < 8) {
            int base = blockIdx.x * 1024;
#pragma unroll
            for (int u = 0; u < 4; ++u) {
                int g = base + tid * 4 + u;      // 0..8191
                int c = g >> 5, k = g & 31;
                W2T[g] = (__bf16)W2[k * 256 + c];
            }
        }
    }
}

// ---------------------------------------------------------------------------
// Kernel 1: Q/K/V projections (M=1024,N=256,K=256).  grid (256,3) x 128 thr.
// Emits Qbf/Kbf row-major bf16 and V transposed bf16 (VT[b][d][s]).
// ---------------------------------------------------------------------------
__global__ void __launch_bounds__(128)
k_proj(const float* q, const float* k, const float* v,
       const __bf16* WqT, const __bf16* WkT, const __bf16* WvT,
       const float* bq, const float* bk, const float* bv,
       __bf16* Qbf, __bf16* Kbf, __bf16* VTb) {
    __shared__ __align__(16) __bf16 As[4][512];
    __shared__ __align__(16) __bf16 Bs[4][512];
    int tid = threadIdx.x, w = tid >> 5, lane = tid & 31;
    int mode = blockIdx.y;

    const float*  X    = (mode == 0) ? q   : (mode == 1) ? k   : v;
    const __bf16* BT   = (mode == 0) ? WqT : (mode == 1) ? WkT : WvT;
    const float*  bias = (mode == 0) ? bq  : (mode == 1) ? bk  : bv;

    int t  = blockIdx.x * 4 + w;          // 64 M-tiles x 16 N-tiles
    int m0 = (t >> 4) * 16;
    int n0 = (t & 15) * 16;

    float acc[8];
    wave_gemm_bfB(X, DDm, m0, BT, DDm, n0, DDm, As[w], Bs[w], lane, acc);

    int nloc = lane & 15, gh = lane >> 4;
    float bb = bias[n0 + nloc];
#pragma unroll
    for (int vv = 0; vv < 8; ++vv) {
        int m = m0 + vv + 8 * gh;
        float val = acc[vv] + bb;
        if (mode == 0)      Qbf[m * DDm + n0 + nloc] = (__bf16)val;
        else if (mode == 1) Kbf[m * DDm + n0 + nloc] = (__bf16)val;
        else {
            int bi = m >> 9, s = m & 511;
            VTb[bi * (DDm * SS) + (n0 + nloc) * SS + s] = (__bf16)val;
        }
    }
}

// ---------------------------------------------------------------------------
// Kernel 2: fused radial expansion -> hdn -> r -> logits -> attn.
// grid 4096 (= B*S*4), 128 threads.  Block = (b, i, 8 j-tiles).
// W1/W2/K-rows staged with TDM; r never hits global memory.
// ---------------------------------------------------------------------------
__global__ void __launch_bounds__(128)
k_attn(const float* weight, const float* mask,
       const __bf16* W1T, const __bf16* W2T, const float* b1, const float* b2,
       const __bf16* Qbf, const __bf16* Kbf, float* attnO) {
    __shared__ __align__(16) __bf16 eeS[512];       // [jj][p]  16x32
    __shared__ __align__(16) __bf16 w1S[2][512];    // == W1T (B-tile layout)
    __shared__ __align__(16) __bf16 hdnS[512];      // [jj][m]  16x32
    __shared__ __align__(16) __bf16 w2S[16][512];   // == W2T (B-tile layout)
    __shared__ __align__(16) __bf16 kSb[16 * 256];  // 16 K-rows (bf16)
    __shared__ float qS[256];
    __shared__ float b2S[256];
    __shared__ float b1S[32];
    __shared__ float dS[16];
    __shared__ float mS[16];

    int tid = threadIdx.x, w = tid >> 5, lane = tid & 31;
    int x   = blockIdx.x;
    int b   = x >> 11;
    int rem = x & 2047;
    int i   = rem >> 2;
    int jt8 = rem & 3;

    const float RSQ   = 0.17677669529663687f;  // 1/sqrt(32)
    const float ESTEP = 2.9f / 31.0f;

    // one-time TDM staging of W1T (1K elems) and W2T (8K elems)
    if (w == 0) {
        tdm_copy_1d(lds_off(&w1S[0][0]), W1T, 1024);
        tdm_copy_1d(lds_off(&w2S[0][0]), W2T, 8192);
    }
    {   // q row, biases
        int c = tid * 2;
        qS[c]      = (float)Qbf[(b * SS + i) * DDm + c];
        qS[c + 1]  = (float)Qbf[(b * SS + i) * DDm + c + 1];
        b2S[c]     = b2[c];
        b2S[c + 1] = b2[c + 1];
    }
    if (tid < 32) b1S[tid] = b1[tid];

    for (int it = 0; it < 8; ++it) {
        int j0 = (jt8 * 8 + it) * 16;
        __syncthreads();   // prior readers of kSb/eeS/hdnS done

        if (w == 0) {      // 16 contiguous K-rows -> one 8KB TDM copy
            tdm_copy_1d(lds_off(kSb), Kbf + (size_t)(b * SS + j0) * DDm, 4096);
        }
        if (tid < 16) {
            dS[tid] = weight[(b * SS + i) * SS + j0 + tid];
            mS[tid] = mask[(b * SS + i) * SS + j0 + tid];
        }
        if (w == 0) wait_tensor0();
        __syncthreads();   // publish kSb (+w1S/w2S on iter 0), dS, mS

        // ee[jj][p] = min(0.3/d, 3) * exp(-d * exps[p])
#pragma unroll
        for (int u = 0; u < 4; ++u) {
            int g = tid + 128 * u;
            int jj = g >> 5, p = g & 31;
            float d = dS[jj];
            float pref = fminf(0.3f / d, 3.0f);
            eeS[g] = (__bf16)(pref * __expf(-d * (0.1f + ESTEP * (float)p)));
        }
        __syncthreads();

        // hdn = elu(ee @ W1 + b1): waves 0,1 produce 16 cols each
        if (w < 2) {
            v16bf afr = load_a_frag(eeS, lane);
            v16bf bfr = load_b_frag(w1S[w], lane);
            V8F acc;
#pragma unroll
            for (int vv = 0; vv < 8; ++vv) acc.f[vv] = 0.0f;
            acc.v = wmma_bf16(afr, bfr, acc.v);
            int nloc = lane & 15, gh = lane >> 4;
            float bbv = b1S[w * 16 + nloc];
#pragma unroll
            for (int vv = 0; vv < 8; ++vv) {
                float xv = acc.f[vv] + bbv;
                xv = (xv > 0.0f) ? xv : (__expf(xv) - 1.0f);
                hdnS[(vv + 8 * gh) * 32 + w * 16 + nloc] = (__bf16)xv;
            }
        }
        __syncthreads();

        // r = hdn @ W2 + b2 (registers only); logits = sum_p r*q*k
        {
            v16bf afr = load_a_frag(hdnS, lane);
            int nloc = lane & 15, gh = lane >> 4;
#pragma unroll
            for (int hh = 0; hh < 2; ++hh) {
                int h = 2 * w + hh;
                float part[8];
#pragma unroll
                for (int vv = 0; vv < 8; ++vv) part[vv] = 0.0f;
#pragma unroll
                for (int tt = 0; tt < 2; ++tt) {
                    int tcol = 2 * h + tt;
                    v16bf bfr = load_b_frag(w2S[tcol], lane);
                    V8F acc;
#pragma unroll
                    for (int vv = 0; vv < 8; ++vv) acc.f[vv] = 0.0f;
                    acc.v = wmma_bf16(afr, bfr, acc.v);
                    int dcol = tcol * 16 + nloc;
                    float qv  = qS[dcol];
                    float b2v = b2S[dcol];
#pragma unroll
                    for (int vv = 0; vv < 8; ++vv) {
                        int jj = vv + 8 * gh;
                        part[vv] += (acc.f[vv] + b2v) * qv
                                    * (float)kSb[jj * 256 + dcol];
                    }
                }
#pragma unroll
                for (int mk = 1; mk <= 8; mk <<= 1) {
#pragma unroll
                    for (int vv = 0; vv < 8; ++vv)
                        part[vv] += __shfl_xor(part[vv], mk, 32);
                }
                if (nloc == 0) {
#pragma unroll
                    for (int vv = 0; vv < 8; ++vv) {
                        int jj = vv + 8 * gh;
                        attnO[(((b * HHn + h) * SS) + i) * SS + j0 + jj] =
                            part[vv] * RSQ * (1.0f - mS[jj]);
                    }
                }
            }
        }
    }
}

// ---------------------------------------------------------------------------
// Kernel 3: ctx = attn @ vh.  16 batched 512x32 GEMMs (K=512).  grid 256.
// ---------------------------------------------------------------------------
__global__ void __launch_bounds__(128)
k_ctx(const float* attn, const __bf16* VTb, float* Ctx) {
    __shared__ __align__(16) __bf16 As[4][512];
    __shared__ __align__(16) __bf16 Bs[4][512];
    int tid = threadIdx.x, w = tid >> 5, lane = tid & 31;

    int t   = blockIdx.x * 4 + w;        // 0..1023
    int bh  = t >> 6;
    int rem = t & 63;
    int mt  = rem >> 1;
    int nt  = rem & 1;
    int b   = bh >> 3, h = bh & 7;

    const float*  A  = attn + (size_t)bh * SS * SS;      // lda = 512
    const __bf16* BT = VTb + (size_t)b * DDm * SS;       // [256][512], ldk = 512
    int n0v = h * DPP + nt * 16;

    float acc[8];
    wave_gemm_bfB(A, SS, mt * 16, BT, SS, n0v, SS, As[w], Bs[w], lane, acc);

    int nloc = lane & 15, gh = lane >> 4;
#pragma unroll
    for (int vv = 0; vv < 8; ++vv) {
        int m = vv + 8 * gh;
        Ctx[(b * SS + mt * 16 + m) * DDm + n0v + nloc] = acc[vv];
    }
}

// ---------------------------------------------------------------------------
// Kernel 4: out = ctx @ Wo + bo.  M=1024, N=256, K=256.  grid 256.
// ---------------------------------------------------------------------------
__global__ void __launch_bounds__(128)
k_out(const float* Ctx, const __bf16* WoT, const float* bo, float* outp) {
    __shared__ __align__(16) __bf16 As[4][512];
    __shared__ __align__(16) __bf16 Bs[4][512];
    int tid = threadIdx.x, w = tid >> 5, lane = tid & 31;

    int t  = blockIdx.x * 4 + w;
    int m0 = (t >> 4) * 16;
    int n0 = (t & 15) * 16;

    float acc[8];
    wave_gemm_bfB(Ctx, DDm, m0, WoT, DDm, n0, DDm, As[w], Bs[w], lane, acc);

    int nloc = lane & 15, gh = lane >> 4;
    float bb = bo[n0 + nloc];
#pragma unroll
    for (int vv = 0; vv < 8; ++vv) {
        int m = vv + 8 * gh;
        outp[(m0 + m) * DDm + n0 + nloc] = acc[vv] + bb;
    }
}

// ---------------------------------------------------------------------------
extern "C" void kernel_launch(void* const* d_in, const int* in_sizes, int n_in,
                              void* d_out, int out_size, void* d_ws, size_t ws_size,
                              hipStream_t stream) {
    const float* v      = (const float*)d_in[0];
    const float* k      = (const float*)d_in[1];
    const float* q      = (const float*)d_in[2];
    const float* weight = (const float*)d_in[3];
    /* d_in[4] = atom_type (unused) */
    const float* mask   = (const float*)d_in[5];
    const float* Wq     = (const float*)d_in[6];
    const float* bq     = (const float*)d_in[7];
    const float* Wk     = (const float*)d_in[8];
    const float* bk     = (const float*)d_in[9];
    const float* Wv     = (const float*)d_in[10];
    const float* bv     = (const float*)d_in[11];
    const float* Wo     = (const float*)d_in[12];
    const float* bo     = (const float*)d_in[13];
    const float* W1     = (const float*)d_in[14];
    const float* b1     = (const float*)d_in[15];
    const float* W2     = (const float*)d_in[16];
    const float* b2     = (const float*)d_in[17];

    float* outO  = (float*)d_out;                       // [B,S,D]
    float* attnO = (float*)d_out + BB * SS * DDm;       // [B,H,S,S]

    // workspace layout (bf16 unless noted), all 16B aligned
    __bf16* WqT = (__bf16*)d_ws;          // 65536
    __bf16* WkT = WqT + 65536;
    __bf16* WvT = WkT + 65536;
    __bf16* WvT2 = WvT;                   // (alias, clarity only)
    __bf16* WoT = WvT + 65536;
    __bf16* W1T = WoT + 65536;            // 1024
    __bf16* W2T = W1T + 1024;             // 8192
    __bf16* Qbf = W2T + 8192;             // 262144
    __bf16* Kbf = Qbf + 262144;
    __bf16* VTb = Kbf + 262144;           // V transposed [B][D][S]
    float*  Ctx = (float*)(VTb + 262144); // 262144 fp32
    (void)WvT2;

    k_prep<<<dim3(64, 6), 256, 0, stream>>>(Wq, Wk, Wv, Wo, W1, W2,
                                            WqT, WkT, WvT, WoT, W1T, W2T);
    k_proj<<<dim3(256, 3), 128, 0, stream>>>(q, k, v, WqT, WkT, WvT,
                                             bq, bk, bv, Qbf, Kbf, VTb);
    k_attn<<<dim3(4096), 128, 0, stream>>>(weight, mask, W1T, W2T, b1, b2,
                                           Qbf, Kbf, attnO);
    k_ctx<<<dim3(256), 128, 0, stream>>>(attnO, VTb, Ctx);
    k_out<<<dim3(256), 128, 0, stream>>>(Ctx, WoT, bo, outO);
}
